// SA_68659347193903
// MI455X (gfx1250) — matrix operand order repaired and checked by reading, not compiled, not run on hardware
//
#include <hip/hip_runtime.h>
#include <hip/hip_bf16.h>
#include <math.h>

typedef _Float16 v16h __attribute__((ext_vector_type(16)));
typedef float    v8f  __attribute__((ext_vector_type(8)));

#define WMMA_F16(a, b, c) \
  __builtin_amdgcn_wmma_f32_16x16x32_f16(false, (a), false, (b), (short)0, (c), false, false)

// ---------------------------------------------------------------------------
// WMMA fragment loaders (CDNA5 16x16x32 f16 layouts, wave32)
// A (16xK tile): lane L holds row M = L%16; elements e<8 -> K = k0 + 8*(L/16) + e,
//                elements e>=8 -> K = k0 + 16 + 8*(L/16) + (e-8)
__device__ __forceinline__ v16h load_a_frag(const _Float16* base, int ld,
                                            int m_base, int k0, int lane) {
  int m = m_base + (lane & 15);
  const _Float16* p = base + (size_t)m * ld + k0 + ((lane >> 4) << 3);
  v16h a;
#pragma unroll
  for (int i = 0; i < 8; ++i) { a[i] = p[i]; a[i + 8] = p[16 + i]; }
  return a;
}

// B (Kx16 tile) where B[k][n] = W[o_base+n][k0+k] (i.e. multiply by W^T):
// lane L holds column N = L%16; elements e<8 -> K = k0 + 8*(L/16) + e,
// elements e>=8 -> K = k0 + 16 + 8*(L/16) + (e-8)
__device__ __forceinline__ v16h load_b_frag_wt(const _Float16* W, int ld,
                                               int o_base, int k0, int lane) {
  int o = o_base + (lane & 15);
  const _Float16* p = W + (size_t)o * ld + k0 + ((lane >> 4) << 3);
  v16h b;
#pragma unroll
  for (int i = 0; i < 8; ++i) { b[i] = p[i]; b[i + 8] = p[16 + i]; }
  return b;
}

// ---------------------------------------------------------------------------
__device__ __forceinline__ float block_reduce_sum(float v, float* red) {
  int t = threadIdx.x;
  red[t] = v;
  __syncthreads();
  for (int off = 128; off >= 1; off >>= 1) {
    if (t < off) red[t] += red[t + off];
    __syncthreads();
  }
  float r = red[0];
  __syncthreads();
  return r;
}

// fp32 -> fp16 convert
__global__ void conv_f16_kernel(const float* __restrict__ in,
                                _Float16* __restrict__ out, int n) {
  int i = blockIdx.x * 256 + threadIdx.x;
  if (i < n) out[i] = (_Float16)in[i];
}

// LayerNorm(x; g_a,g_b) then u = xn.w1, t = xn.w2  (one block per token)
__global__ __launch_bounds__(256) void ln_ut_kernel(
    const float* __restrict__ x, const float* __restrict__ ga,
    const float* __restrict__ gb, const float* __restrict__ fcw,
    float* __restrict__ u, float* __restrict__ t, int H) {
  __shared__ float red[256];
  int tok = blockIdx.x;
  const float* xr = x + (size_t)tok * H;
  float s = 0.f;
  for (int i = threadIdx.x; i < H; i += 256) s += xr[i];
  float mean = block_reduce_sum(s, red) / (float)H;
  float vs = 0.f;
  for (int i = threadIdx.x; i < H; i += 256) { float d = xr[i] - mean; vs += d * d; }
  float sd = sqrtf(block_reduce_sum(vs, red) / (float)(H - 1));
  float inv = 1.f / (sd + 1e-6f);
  float su = 0.f, st = 0.f;
  for (int i = threadIdx.x; i < H; i += 256) {
    float xn = ga[i] * (xr[i] - mean) * inv + gb[i];
    su += xn * fcw[i];
    st += xn * fcw[H + i];
  }
  float U = block_reduce_sum(su, red);
  float T = block_reduce_sum(st, red);
  if (threadIdx.x == 0) { u[tok] = U; t[tok] = T; }
}

// per-batch inclusive cumsum of t -> p  (one block per b, S threads)
__global__ void cumsum_kernel(const float* __restrict__ t,
                              float* __restrict__ p, int S) {
  __shared__ float buf[128];
  int b = blockIdx.x, j = threadIdx.x;
  buf[j] = t[b * S + j];
  __syncthreads();
  for (int off = 1; off < S; off <<= 1) {
    float v = (j >= off) ? buf[j - off] : 0.f;
    __syncthreads();
    buf[j] += v;
    __syncthreads();
  }
  p[b * S + j] = buf[j];
}

// Mij + G (one block per (b,i), S threads = j)
__global__ void mij_g_kernel(const float* __restrict__ u,
                             const float* __restrict__ p,
                             const float* __restrict__ prior,
                             const float* __restrict__ fcb,
                             float* __restrict__ Mout,
                             float* __restrict__ Gout, int S) {
  __shared__ float buf[128];
  int b = blockIdx.x / S, i = blockIdx.x % S, j = threadIdx.x;
  float pr = prior[0], fb = fcb[0];
  float p0 = (i > 0) ? p[b * S + i - 1] : 0.f;
  float logit = (j >= i)
      ? (u[b * S + j] + (p[b * S + j] - p0) / (float)(j - i + 1) + fb)
      : fb;
  float mij = pr + (1.f - pr) * (1.f / (1.f + expf(-logit)));
  Mout[((size_t)b * S + i) * S + j] = mij;
  buf[j] = (j >= i) ? mij : 1.f;
  __syncthreads();
  for (int off = 1; off < S; off <<= 1) {
    float v = (j >= off) ? buf[j - off] : 1.f;
    __syncthreads();
    buf[j] *= v;
    __syncthreads();
  }
  float g = buf[j];
  size_t rowi = ((size_t)b * S + i) * S;
  if (j == i) {
    Gout[rowi + i] = 1.f;
  } else if (j > i) {
    Gout[rowi + j] = g;
    Gout[((size_t)b * S + j) * S + i] = g;  // symmetric lower part
  }
}

// Generic WMMA GEMM: out[m,o] = relu?( A[m,:] . W[o,:] + bias[o] )
// one wave per 32(M) x 64(O) tile; K unrolled x2 with ping-pong fragment
// buffers (no register rotation). Requires K % 64 == 0.
__global__ __launch_bounds__(256) void gemm_f16_kernel(
    const _Float16* __restrict__ A, const _Float16* __restrict__ W,
    const float* __restrict__ bias, float* __restrict__ outF,
    _Float16* __restrict__ outH, int M, int K, int O, int relu) {
  int lane = threadIdx.x & 31;
  int w = blockIdx.x * 8 + (threadIdx.x >> 5);
  int mtiles = M >> 5;  // 32-row tiles
  int mt = w % mtiles;
  int nb = w / mtiles;
  int m_base = mt << 5;
  int o_base = nb << 6;
  if (o_base >= O) return;

  v8f acc[2][4] = {};
  // ping buffer (set P) for k = 0
  v16h a0P = load_a_frag(A, K, m_base, 0, lane);
  v16h a1P = load_a_frag(A, K, m_base + 16, 0, lane);
  v16h bP[4];
#pragma unroll
  for (int t = 0; t < 4; ++t) bP[t] = load_b_frag_wt(W, K, o_base + t * 16, 0, lane);
  v16h a0Q, a1Q, bQ[4];  // pong buffer (set Q)

  for (int k0 = 0; k0 < K; k0 += 64) {
    // prefetch ahead into L2 (global_prefetch_b8)
    if (k0 + 128 < K) {
      __builtin_prefetch(A + (size_t)(m_base + (lane & 15)) * K + k0 + 128, 0, 1);
      __builtin_prefetch(W + (size_t)(o_base + (lane & 15)) * K + k0 + 128, 0, 1);
    }
    // load Q for k0+32, then compute with P
    int k1 = k0 + 32;
    a0Q = load_a_frag(A, K, m_base, k1, lane);
    a1Q = load_a_frag(A, K, m_base + 16, k1, lane);
#pragma unroll
    for (int t = 0; t < 4; ++t) bQ[t] = load_b_frag_wt(W, K, o_base + t * 16, k1, lane);
#pragma unroll
    for (int t = 0; t < 4; ++t) {
      acc[0][t] = WMMA_F16(a0P, bP[t], acc[0][t]);
      acc[1][t] = WMMA_F16(a1P, bP[t], acc[1][t]);
    }
    // load P for k0+64 (dummy reload of k=0 on final pass), compute with Q
    int k2 = (k0 + 64 < K) ? (k0 + 64) : 0;
    a0P = load_a_frag(A, K, m_base, k2, lane);
    a1P = load_a_frag(A, K, m_base + 16, k2, lane);
#pragma unroll
    for (int t = 0; t < 4; ++t) bP[t] = load_b_frag_wt(W, K, o_base + t * 16, k2, lane);
#pragma unroll
    for (int t = 0; t < 4; ++t) {
      acc[0][t] = WMMA_F16(a0Q, bQ[t], acc[0][t]);
      acc[1][t] = WMMA_F16(a1Q, bQ[t], acc[1][t]);
    }
  }

#pragma unroll
  for (int half = 0; half < 2; ++half) {
#pragma unroll
    for (int t = 0; t < 4; ++t) {
      int o = o_base + t * 16 + (lane & 15);
      float bv = bias ? bias[o] : 0.f;
#pragma unroll
      for (int r = 0; r < 8; ++r) {
        int m = m_base + half * 16 + r + ((lane >> 4) << 3);
        float v = acc[half][t][r] + bv;
        if (relu) v = fmaxf(v, 0.f);
        size_t idx = (size_t)m * O + o;
        if (outF) outF[idx] = v;
        if (outH) outH[idx] = (_Float16)v;
      }
    }
  }
}

// Fused attention per (b,h): scores = q k^T/8 (+mask) -> softmax -> *G -> ctx = att v
// Assumes S == 128, dh == 64 (reference shapes).
#define SS 136  // padded LDS row stride (halfs) to break bank conflicts
__global__ __launch_bounds__(256) void attn_kernel(
    const _Float16* __restrict__ qh, const _Float16* __restrict__ kh,
    const _Float16* __restrict__ vh, const float* __restrict__ G,
    const unsigned char* __restrict__ mask, _Float16* __restrict__ ctxh,
    int S, int H, int nh) {
  __shared__ _Float16 s_sc[128 * SS];  // scores (f16), then att (f16) in-place
  __shared__ _Float16 s_vt[64 * SS];   // v transposed: vt[d][key]
  __shared__ float s_mask[128];
  const int dh = 64;
  int b = blockIdx.x / nh, h = blockIdx.x % nh;
  int lane = threadIdx.x & 31;
  int wave = threadIdx.x >> 5;
  const _Float16* qb = qh + ((size_t)b * S) * H + h * dh;
  const _Float16* kb = kh + ((size_t)b * S) * H + h * dh;
  const _Float16* vb = vh + ((size_t)b * S) * H + h * dh;

  if (threadIdx.x < 128) s_mask[threadIdx.x] = mask[b * S + threadIdx.x] ? 1.f : 0.f;
  // stage v^T into LDS (contiguous B-fragments for phase 3)
  for (int idx = threadIdx.x; idx < 128 * 64; idx += 256) {
    int key = idx >> 6;
    int d = idx & 63;
    s_vt[d * SS + key] = vb[(size_t)key * H + d];
  }
  __syncthreads();

  // phase 1: scores, 8x8 grid of 16x16 tiles, 8 tiles per wave; dh fully unrolled
  for (int tile = wave; tile < 64; tile += 8) {
    int mt = tile >> 3, nt = tile & 7;
    v16h aq0 = load_a_frag(qb, H, mt * 16, 0, lane);
    v16h bk0 = load_b_frag_wt(kb, H, nt * 16, 0, lane);  // B[k][n] = K[n][d]
    v16h aq1 = load_a_frag(qb, H, mt * 16, 32, lane);
    v16h bk1 = load_b_frag_wt(kb, H, nt * 16, 32, lane);
    v8f acc = {};
    acc = WMMA_F16(aq0, bk0, acc);
    acc = WMMA_F16(aq1, bk1, acc);
    int n = nt * 16 + (lane & 15);
    float msk = s_mask[n];
#pragma unroll
    for (int r = 0; r < 8; ++r) {
      int m = mt * 16 + r + ((lane >> 4) << 3);
      float v = (msk > 0.5f) ? -30000.f : acc[r] * 0.125f;  // 1/sqrt(64)
      s_sc[m * SS + n] = (_Float16)v;
    }
  }
  __syncthreads();

  // phase 2: per-row softmax * G, in place (row tid owns its row)
  if (threadIdx.x < 128) {
    int row = threadIdx.x;
    float mx = -3.4e38f;
    for (int c = 0; c < 128; ++c) mx = fmaxf(mx, (float)s_sc[row * SS + c]);
    float sum = 0.f;
    for (int c = 0; c < 128; ++c) sum += expf((float)s_sc[row * SS + c] - mx);
    float inv = 1.f / sum;
    const float* Gr = G + ((size_t)b * S + row) * S;
    for (int c = 0; c < 128; ++c) {
      float v = expf((float)s_sc[row * SS + c] - mx) * inv * Gr[c];
      s_sc[row * SS + c] = (_Float16)v;
    }
  }
  __syncthreads();

  // phase 3: ctx = att @ v, 8x4 grid of 16x16 tiles, 4 tiles per wave
  for (int tile = wave; tile < 32; tile += 8) {
    int mt = tile >> 2, nt = tile & 3;
    v8f acc = {};
#pragma unroll
    for (int k0 = 0; k0 < 128; k0 += 32) {
      v16h a = load_a_frag(s_sc, SS, mt * 16, k0, lane);
      // B[k][n] = v[k][d=n] = vt[n][k]  -> contiguous LDS reads
      v16h bb = load_b_frag_wt(s_vt, SS, nt * 16, k0, lane);
      acc = WMMA_F16(a, bb, acc);
    }
    int n = nt * 16 + (lane & 15);
#pragma unroll
    for (int r = 0; r < 8; ++r) {
      int m = mt * 16 + r + ((lane >> 4) << 3);
      ctxh[((size_t)b * S + m) * H + h * dh + n] = (_Float16)acc[r];
    }
  }
}

// out = LN(xa + xb; gamma, beta), optional f32 + f16 outputs (block per token)
__global__ __launch_bounds__(256) void ln_res_kernel(
    const float* __restrict__ xa, const float* __restrict__ xb,
    const float* __restrict__ gamma, const float* __restrict__ beta,
    float* __restrict__ outF, _Float16* __restrict__ outH, int H) {
  __shared__ float red[256];
  int tok = blockIdx.x;
  const float* pa = xa + (size_t)tok * H;
  const float* pb = xb + (size_t)tok * H;
  float s = 0.f;
  for (int i = threadIdx.x; i < H; i += 256) s += pa[i] + pb[i];
  float mean = block_reduce_sum(s, red) / (float)H;
  float vs = 0.f;
  for (int i = threadIdx.x; i < H; i += 256) {
    float d = pa[i] + pb[i] - mean;
    vs += d * d;
  }
  float sd = sqrtf(block_reduce_sum(vs, red) / (float)(H - 1));
  float inv = 1.f / (sd + 1e-6f);
  for (int i = threadIdx.x; i < H; i += 256) {
    float y = gamma[i] * (pa[i] + pb[i] - mean) * inv + beta[i];
    size_t idx = (size_t)tok * H + i;
    if (outF) outF[idx] = y;
    if (outH) outH[idx] = (_Float16)y;
  }
}

// ---------------------------------------------------------------------------
extern "C" void kernel_launch(void* const* d_in, const int* in_sizes, int n_in,
                              void* d_out, int out_size, void* d_ws,
                              size_t ws_size, hipStream_t stream) {
  const float* x = (const float*)d_in[0];
  const unsigned char* xmask = (const unsigned char*)d_in[1];  // bool mask
  const float* prior = (const float*)d_in[2];
  const float* g_a = (const float*)d_in[3];
  const float* g_b = (const float*)d_in[4];
  const float* fc_w = (const float*)d_in[5];
  const float* fc_b = (const float*)d_in[6];
  const float* wq = (const float*)d_in[7];
  const float* bq = (const float*)d_in[8];
  const float* wk = (const float*)d_in[9];
  const float* bk = (const float*)d_in[10];
  const float* wv = (const float*)d_in[11];
  const float* bv = (const float*)d_in[12];
  const float* wm = (const float*)d_in[13];
  const float* bm = (const float*)d_in[14];
  const float* n1_a = (const float*)d_in[15];
  const float* n1_b = (const float*)d_in[16];
  const float* n2_a = (const float*)d_in[17];
  const float* n2_b = (const float*)d_in[18];
  const float* w_ff1 = (const float*)d_in[19];
  const float* b_ff1 = (const float*)d_in[20];
  const float* w_ff2 = (const float*)d_in[21];
  const float* b_ff2 = (const float*)d_in[22];

  const int H = in_sizes[3];       // 512
  const int FF = in_sizes[20];     // 2048
  const int BS = in_sizes[0] / H;  // B*S = 1024
  const int S = 128;
  const int B = BS / S;
  const int NH = 8;

  float* outX2 = (float*)d_out;
  float* outG = outX2 + (size_t)BS * H;
  float* outM = outG + (size_t)B * S * S;

  // workspace bump allocator
  char* ws = (char*)d_ws;
  size_t off = 0;
  auto alloc = [&](size_t bytes) -> void* {
    void* p = ws + off;
    off = (off + bytes + 255) & ~(size_t)255;
    return p;
  };
  _Float16* xh    = (_Float16*)alloc((size_t)BS * H * 2);
  _Float16* wqh   = (_Float16*)alloc((size_t)H * H * 2);
  _Float16* wkh   = (_Float16*)alloc((size_t)H * H * 2);
  _Float16* wvh   = (_Float16*)alloc((size_t)H * H * 2);
  _Float16* wmh   = (_Float16*)alloc((size_t)H * H * 2);
  _Float16* wff1h = (_Float16*)alloc((size_t)FF * H * 2);
  _Float16* wff2h = (_Float16*)alloc((size_t)H * FF * 2);
  _Float16* qh    = (_Float16*)alloc((size_t)BS * H * 2);
  _Float16* kh    = (_Float16*)alloc((size_t)BS * H * 2);
  _Float16* vh    = (_Float16*)alloc((size_t)BS * H * 2);
  _Float16* ctxh  = (_Float16*)alloc((size_t)BS * H * 2);
  _Float16* x1h   = (_Float16*)alloc((size_t)BS * H * 2);
  _Float16* ff1h  = (_Float16*)alloc((size_t)BS * FF * 2);
  float* atted = (float*)alloc((size_t)BS * H * 4);
  float* ffout = (float*)alloc((size_t)BS * H * 4);
  float* x1f   = (float*)alloc((size_t)BS * H * 4);
  float* u_buf = (float*)alloc((size_t)BS * 4);
  float* t_buf = (float*)alloc((size_t)BS * 4);
  float* p_buf = (float*)alloc((size_t)BS * 4);

  auto conv = [&](const float* src, _Float16* dst, int n) {
    conv_f16_kernel<<<(n + 255) / 256, 256, 0, stream>>>(src, dst, n);
  };
  conv(x, xh, BS * H);
  conv(wq, wqh, H * H);
  conv(wk, wkh, H * H);
  conv(wv, wvh, H * H);
  conv(wm, wmh, H * H);
  conv(w_ff1, wff1h, FF * H);
  conv(w_ff2, wff2h, H * FF);

  // group-attention scalar path: u,t -> cumsum p -> Mij,G
  ln_ut_kernel<<<BS, 256, 0, stream>>>(x, g_a, g_b, fc_w, u_buf, t_buf, H);
  cumsum_kernel<<<B, S, 0, stream>>>(t_buf, p_buf, S);
  mij_g_kernel<<<B * S, S, 0, stream>>>(u_buf, p_buf, prior, fc_b, outM, outG, S);

  auto gemm = [&](const _Float16* A, const _Float16* W, const float* bias,
                  float* oF, _Float16* oH, int M, int K, int O, int relu) {
    int waves = (M / 32) * (O / 64);
    gemm_f16_kernel<<<waves / 8, 256, 0, stream>>>(A, W, bias, oF, oH, M, K, O, relu);
  };

  // q,k,v projections (f16 outputs feed WMMA attention)
  gemm(xh, wqh, bq, nullptr, qh, BS, H, H, 0);
  gemm(xh, wkh, bk, nullptr, kh, BS, H, H, 0);
  gemm(xh, wvh, bv, nullptr, vh, BS, H, H, 0);

  // fused attention (reads G from d_out)
  attn_kernel<<<B * NH, 256, 0, stream>>>(qh, kh, vh, outG, xmask, ctxh, S, H, NH);

  // output projection + residual LN1
  gemm(ctxh, wmh, bm, atted, nullptr, BS, H, H, 0);
  ln_res_kernel<<<BS, 256, 0, stream>>>(x, atted, n1_a, n1_b, x1f, x1h, H);

  // FFN
  gemm(x1h, wff1h, b_ff1, nullptr, ff1h, BS, H, FF, 1);
  gemm(ff1h, wff2h, b_ff2, ffout, nullptr, BS, FF, H, 0);

  // residual LN2 -> x2 output
  ln_res_kernel<<<BS, 256, 0, stream>>>(x1f, ffout, n2_a, n2_b, outX2, nullptr, H);
}
